// Network_51445118271910
// MI455X (gfx1250) — matrix operand order repaired and checked by reading, not compiled, not run on hardware
//
#include <hip/hip_runtime.h>
#include <math.h>

// Shapes (batch row 0 only — output depends only on probs[0])
#define SEQ   256
#define HID   512
#define NH    8
#define DK    64      // HID / NH
#define OUTN  32000

typedef __attribute__((ext_vector_type(16))) __bf16 v16bf;
typedef __attribute__((ext_vector_type(8)))  __bf16 v8bf;
typedef __attribute__((ext_vector_type(8)))  float  v8f;

static __device__ __forceinline__ v8f wmma_bf16(v16bf a, v16bf b, v8f c) {
  // (neg_a, A, neg_b, B, c_mod, C, reuse_a, reuse_b)
  return __builtin_amdgcn_wmma_f32_16x16x32_bf16(false, a, false, b, (short)0, c, false, false);
}

static __device__ __forceinline__ v16bf frag2(const __bf16* p0, const __bf16* p1) {
  return __builtin_shufflevector(*(const v8bf*)p0, *(const v8bf*)(p1),
                                 0,1,2,3,4,5,6,7,8,9,10,11,12,13,14,15);
}

// ------------- weight conversion: wq/wk/wv (2 layers) -> bf16 TRANSPOSED --------
// Wt[(l*3+o)][n*HID + k] = w[(l)][k*HID + n]; makes WMMA B-fragments contiguous.
__global__ void k_convert_w(const float* __restrict__ wq, const float* __restrict__ wk,
                            const float* __restrict__ wv, __bf16* __restrict__ wt) {
  int idx   = blockIdx.x * blockDim.x + threadIdx.x;   // 6*262144 threads
  int which = idx >> 18;                               // (layer*3 + {q,k,v})
  int r     = idx & 262143;                            // source offset = k*HID + n
  int k = r >> 9, n = r & 511;
  int l = which / 3, o = which % 3;
  const float* src = (o == 0) ? wq : (o == 1) ? wk : wv;
  wt[(size_t)which * HID * HID + (size_t)n * HID + k] = (__bf16)src[(size_t)l * HID * HID + r];
}

// ---------------- embedding + positional encoding (batch row 0) ----------------
__global__ void k_embed(const int* __restrict__ x, const float* __restrict__ emb,
                        __bf16* __restrict__ abf) {
  int idx = blockIdx.x * blockDim.x + threadIdx.x;     // 256*512
  int s = idx >> 9, c = idx & 511;
  int tok = x[s];                                      // x[0, s]
  float v = emb[(size_t)tok * HID + c];
  float expo = (float)(c & ~1) / (float)HID;           // 2*(c/2)/H
  float ang  = (float)s * powf(10000.0f, -expo);
  float pe   = (c & 1) ? cosf(ang) : sinf(ang);
  abf[idx] = (__bf16)(v + pe);
}

// ------------- QKV projections: (256x512) @ (512x512) via WMMA bf16 -------------
// 384 waves: out(3) x mt(16) x ng(8). Each wave: 4 N-tiles (4 accs), A reused.
// Q,K stored row-major (S x H); V stored TRANSPOSED (H x S) for the P@V kernel.
__global__ void k_qkv(const __bf16* __restrict__ A, const __bf16* __restrict__ Wt,
                      __bf16* __restrict__ Q, __bf16* __restrict__ K, __bf16* __restrict__ Vt) {
  int wave = blockIdx.x * (blockDim.x >> 5) + (threadIdx.x >> 5);
  int lane = threadIdx.x & 31;
  int out  = wave >> 7;                 // /128
  int rem  = wave & 127;
  int mt = rem >> 3, ng = rem & 7;      // n0 = ng*64
  const __bf16* W = Wt + (size_t)out * HID * HID;
  int m = lane & 15, half = lane >> 4;

  v8f acc[4] = {};
  for (int kb = 0; kb < HID; kb += 32) {
    const __bf16* ap = A + (size_t)(mt * 16 + m) * HID + kb + half * 16;
    v16bf a = frag2(ap, ap + 8);
    #pragma unroll
    for (int t = 0; t < 4; ++t) {
      const __bf16* bp = W + (size_t)(ng * 64 + t * 16 + m) * HID + kb + half * 16;
      v16bf b = frag2(bp, bp + 8);
      acc[t] = wmma_bf16(a, b, acc[t]);
    }
  }
  #pragma unroll
  for (int t = 0; t < 4; ++t) {
    #pragma unroll
    for (int r = 0; r < 8; ++r) {
      int row = mt * 16 + half * 8 + r;
      int col = ng * 64 + t * 16 + m;
      __bf16 v = (__bf16)acc[t][r];
      if (out == 0)      Q[(size_t)row * HID + col] = v;
      else if (out == 1) K[(size_t)row * HID + col] = v;
      else               Vt[(size_t)col * SEQ + row] = v;   // transposed
    }
  }
}

// ------------- scores: S = (Q Kt) * 1/sqrt(dk), per head, via WMMA --------------
// 512 waves: head(8) x mt(16) x ng(4). Each wave: 4 N-tiles, A reused.
__global__ void k_scores(const __bf16* __restrict__ Q, const __bf16* __restrict__ K,
                         float* __restrict__ S) {
  int wave = blockIdx.x * (blockDim.x >> 5) + (threadIdx.x >> 5);
  int lane = threadIdx.x & 31;
  int head = wave >> 6;
  int rem  = wave & 63;
  int mt = rem >> 2, ng = rem & 3;      // n0 = ng*64
  int m = lane & 15, half = lane >> 4;

  v8f acc[4] = {};
  #pragma unroll
  for (int kb = 0; kb < DK; kb += 32) {
    const __bf16* ap = Q + (size_t)(mt * 16 + m) * HID + head * DK + kb + half * 16;
    v16bf a = frag2(ap, ap + 8);
    #pragma unroll
    for (int t = 0; t < 4; ++t) {
      // B = K^T: 16 consecutive K-dim elements are contiguous in K's row
      const __bf16* bp = K + (size_t)(ng * 64 + t * 16 + m) * HID + head * DK + kb + half * 16;
      v16bf b = frag2(bp, bp + 8);
      acc[t] = wmma_bf16(a, b, acc[t]);
    }
  }
  #pragma unroll
  for (int t = 0; t < 4; ++t) {
    #pragma unroll
    for (int r = 0; r < 8; ++r) {
      int row = mt * 16 + half * 8 + r;
      int col = ng * 64 + t * 16 + m;
      S[(size_t)head * SEQ * SEQ + (size_t)row * SEQ + col] = acc[t][r] * 0.125f;
    }
  }
}

// ---------------- row softmax over 256, output bf16 P ---------------------------
__global__ void k_softmax_rows(const float* __restrict__ S, __bf16* __restrict__ P) {
  __shared__ float red[256];
  int row = blockIdx.x;          // head*256 + s
  int tid = threadIdx.x;
  float v = S[(size_t)row * SEQ + tid];
  red[tid] = v; __syncthreads();
  for (int off = 128; off > 0; off >>= 1) {
    if (tid < off) red[tid] = fmaxf(red[tid], red[tid + off]);
    __syncthreads();
  }
  float mx = red[0]; __syncthreads();
  float e = expf(v - mx);
  red[tid] = e; __syncthreads();
  for (int off = 128; off > 0; off >>= 1) {
    if (tid < off) red[tid] += red[tid + off];
    __syncthreads();
  }
  float s = red[0];
  P[(size_t)row * SEQ + tid] = (__bf16)(e / s);
}

// ------------- attention output: O = P @ V (V pre-transposed), via WMMA ---------
// 128 waves: head(8) x mt(16). Each wave: all 4 N-tiles of dk. f32 + bf16 out.
__global__ void k_pv(const __bf16* __restrict__ P, const __bf16* __restrict__ Vt,
                     float* __restrict__ Of, __bf16* __restrict__ Ob) {
  int wave = blockIdx.x * (blockDim.x >> 5) + (threadIdx.x >> 5);
  int lane = threadIdx.x & 31;
  int head = wave >> 4;
  int mt   = wave & 15;
  int m = lane & 15, half = lane >> 4;

  v8f acc[4] = {};
  for (int kb = 0; kb < SEQ; kb += 32) {
    const __bf16* ap = P + (size_t)head * SEQ * SEQ + (size_t)(mt * 16 + m) * SEQ + kb + half * 16;
    v16bf a = frag2(ap, ap + 8);
    #pragma unroll
    for (int t = 0; t < 4; ++t) {
      const __bf16* bp = Vt + (size_t)(head * DK + t * 16 + m) * SEQ + kb + half * 16;
      v16bf b = frag2(bp, bp + 8);
      acc[t] = wmma_bf16(a, b, acc[t]);
    }
  }
  #pragma unroll
  for (int t = 0; t < 4; ++t) {
    #pragma unroll
    for (int r = 0; r < 8; ++r) {
      int row = mt * 16 + half * 8 + r;
      int col = head * DK + t * 16 + m;
      float f = acc[t][r];
      Of[(size_t)row * HID + col] = f;
      Ob[(size_t)row * HID + col] = (__bf16)f;
    }
  }
}

// ------------- head matvec 1: streams w1 (256 MB), per-block partials -----------
__global__ void k_mv1(const float* __restrict__ flat, const float* __restrict__ w1,
                      float* __restrict__ partial) {
  int b   = blockIdx.x;          // 256 blocks, 512 rows each
  int tid = threadIdx.x;         // 256 threads, 2 cols each
  int r0  = b * 512;
  float acc0 = 0.f, acc1 = 0.f;
  for (int r = 0; r < 512; ++r) {
    float xv = flat[r0 + r];
    const float* wr = w1 + (size_t)(r0 + r) * HID;
    __builtin_prefetch(wr + 8 * HID + tid, 0, 1);  // global_prefetch_b8 on the w1 stream
    acc0 += xv * wr[tid];
    acc1 += xv * wr[tid + 256];
  }
  partial[(size_t)b * HID + tid]       = acc0;
  partial[(size_t)b * HID + tid + 256] = acc1;
}

__global__ void k_reduce_h(const float* __restrict__ partial, const float* __restrict__ b1,
                           float* __restrict__ h) {
  int j = blockIdx.x * blockDim.x + threadIdx.x;   // 512
  float s = 0.f;
  for (int b = 0; b < 256; ++b) s += partial[(size_t)b * HID + j];
  h[j] = fmaxf(s + b1[j], 0.0f);                   // bias + ReLU
}

// ------------- head matvec 2: streams w2 (64 MB), float4 coalesced --------------
__global__ void k_mv2(const float* __restrict__ h, const float* __restrict__ w2,
                      const float* __restrict__ b2, float* __restrict__ g) {
  int t = blockIdx.x * blockDim.x + threadIdx.x;   // 8192, guard to 8000
  if (t >= OUTN / 4) return;
  float ax = 0.f, ay = 0.f, az = 0.f, aw = 0.f;
  const float* base = w2 + 4 * t;
  for (int j = 0; j < HID; ++j) {
    float hv = h[j];
    float4 w = *(const float4*)(base + (size_t)j * OUTN);
    ax += hv * w.x; ay += hv * w.y; az += hv * w.z; aw += hv * w.w;
  }
  int k = 4 * t;
  g[k]     = ax + b2[k];
  g[k + 1] = ay + b2[k + 1];
  g[k + 2] = az + b2[k + 2];
  g[k + 3] = aw + b2[k + 3];
}

// ------------- double softmax over 32000 + trailing 0.0 loss --------------------
__global__ void k_final(const float* __restrict__ g, float* __restrict__ out) {
  __shared__ float red[1024];
  int tid = threadIdx.x;   // 1024 threads, one block
  float mx = -3.402823466e38f;
  for (int k = tid; k < OUTN; k += 1024) mx = fmaxf(mx, g[k]);
  red[tid] = mx; __syncthreads();
  for (int off = 512; off > 0; off >>= 1) {
    if (tid < off) red[tid] = fmaxf(red[tid], red[tid + off]);
    __syncthreads();
  }
  float m1 = red[0]; __syncthreads();
  float sm = 0.f;
  for (int k = tid; k < OUTN; k += 1024) sm += expf(g[k] - m1);
  red[tid] = sm; __syncthreads();
  for (int off = 512; off > 0; off >>= 1) {
    if (tid < off) red[tid] += red[tid + off];
    __syncthreads();
  }
  float s1 = red[0]; __syncthreads();
  float inv_s1 = 1.0f / s1;
  // second softmax: max of p1 = exp(0)/s1 = 1/s1 (softmax is monotone in g)
  float m2 = inv_s1;
  float s2l = 0.f;
  for (int k = tid; k < OUTN; k += 1024) {
    float p1 = expf(g[k] - m1) * inv_s1;
    s2l += expf(p1 - m2);
  }
  red[tid] = s2l; __syncthreads();
  for (int off = 512; off > 0; off >>= 1) {
    if (tid < off) red[tid] += red[tid + off];
    __syncthreads();
  }
  float s2 = red[0]; __syncthreads();
  for (int k = tid; k < OUTN; k += 1024) {
    float p1 = expf(g[k] - m1) * inv_s1;
    out[k] = expf(p1 - m2) / s2;
  }
  if (tid == 0) out[OUTN] = 0.0f;   // reference's constant loss
}

extern "C" void kernel_launch(void* const* d_in, const int* in_sizes, int n_in,
                              void* d_out, int out_size, void* d_ws, size_t ws_size,
                              hipStream_t stream) {
  (void)in_sizes; (void)n_in; (void)out_size; (void)ws_size;
  const int*   x   = (const int*)d_in[0];
  const float* emb = (const float*)d_in[1];
  const float* wq  = (const float*)d_in[2];
  const float* wk  = (const float*)d_in[3];
  const float* wv  = (const float*)d_in[4];
  const float* w1  = (const float*)d_in[5];
  const float* b1  = (const float*)d_in[6];
  const float* w2  = (const float*)d_in[7];
  const float* b2  = (const float*)d_in[8];
  float* out = (float*)d_out;

  // workspace layout (bytes)
  char* ws = (char*)d_ws;
  size_t o = 0;
  __bf16* Wt  = (__bf16*)(ws + o); o += (size_t)6 * HID * HID * 2;   // 3 MB (transposed bf16)
  __bf16* A0  = (__bf16*)(ws + o); o += (size_t)SEQ * HID * 2;
  __bf16* A1  = (__bf16*)(ws + o); o += (size_t)SEQ * HID * 2;
  __bf16* Qb  = (__bf16*)(ws + o); o += (size_t)SEQ * HID * 2;
  __bf16* Kb  = (__bf16*)(ws + o); o += (size_t)SEQ * HID * 2;
  __bf16* Vt  = (__bf16*)(ws + o); o += (size_t)SEQ * HID * 2;       // V transposed (H x S)
  float*  S   = (float*)(ws + o);  o += (size_t)NH * SEQ * SEQ * 4;  // 2 MB
  __bf16* P   = (__bf16*)(ws + o); o += (size_t)NH * SEQ * SEQ * 2;  // 1 MB
  float*  O0  = (float*)(ws + o);  o += (size_t)SEQ * HID * 4;
  float*  flat= (float*)(ws + o);  o += (size_t)SEQ * HID * 4;
  float*  part= (float*)(ws + o);  o += (size_t)256 * HID * 4;
  float*  h   = (float*)(ws + o);  o += (size_t)HID * 4;
  float*  g   = (float*)(ws + o);  o += (size_t)OUTN * 4;

  // weight conversion (transposed) + embedding
  k_convert_w<<<6144, 256, 0, stream>>>(wq, wk, wv, Wt);
  k_embed<<<512, 256, 0, stream>>>(x, emb, A0);

  // layer 0
  k_qkv<<<48, 256, 0, stream>>>(A0, Wt, Qb, Kb, Vt);
  k_scores<<<64, 256, 0, stream>>>(Qb, Kb, S);
  k_softmax_rows<<<2048, 256, 0, stream>>>(S, P);
  k_pv<<<16, 256, 0, stream>>>(P, Vt, O0, A1);

  // layer 1
  k_qkv<<<48, 256, 0, stream>>>(A1, Wt + (size_t)3 * HID * HID, Qb, Kb, Vt);
  k_scores<<<64, 256, 0, stream>>>(Qb, Kb, S);
  k_softmax_rows<<<2048, 256, 0, stream>>>(S, P);
  k_pv<<<16, 256, 0, stream>>>(P, Vt, flat, A0);

  // MLP head (bandwidth-bound: streams w1 then w2)
  k_mv1<<<256, 256, 0, stream>>>(flat, w1, part);
  k_reduce_h<<<2, 256, 0, stream>>>(part, b1, h);
  k_mv2<<<32, 256, 0, stream>>>(h, w2, b2, g);
  k_final<<<1, 1024, 0, stream>>>(g, out);
}
